// Net_28827820491044
// MI455X (gfx1250) — matrix operand order repaired and checked by reading, not compiled, not run on hardware
//
#include <hip/hip_runtime.h>
#include <math.h>

// ---------------- problem constants (from reference) ----------------
constexpr int BB  = 8;
constexpr int NN  = 2048;
constexpr int NCLS = 16;
constexpr int OUTC = 50;

// ---------------- bf16 helpers (manual, RNE) ----------------
__device__ __forceinline__ unsigned short f2bf(float f) {
  unsigned int u = __float_as_uint(f);
  u = (u + 0x7FFFu + ((u >> 16) & 1u)) >> 16;
  return (unsigned short)u;
}
__device__ __forceinline__ float bf2f(unsigned short h) {
  return __uint_as_float(((unsigned int)h) << 16);
}

typedef __attribute__((ext_vector_type(16))) __bf16        v16bf;
typedef __attribute__((ext_vector_type(8)))  float         v8f;
typedef __attribute__((ext_vector_type(8)))  unsigned short ushort8;

union BfFrag { v16bf bf; ushort8 u[2]; };

// ---- CDNA5 async global->LDS copy (ASYNCcnt path), 16B per lane ----
// Generic LDS pointers on gfx1250 carry the LDS byte offset in their low 32
// bits (aperture truncation), so a plain cast yields the VDST operand.
__device__ __forceinline__ void async_ld_b128(const void* g, void* l) {
  unsigned     lds = (unsigned)(size_t)l;
  unsigned long long ga = (unsigned long long)(size_t)g;
  asm volatile("global_load_async_to_lds_b128 %0, %1, off"
               :: "v"(lds), "v"(ga) : "memory");
}
__device__ __forceinline__ void wait_async0() {
  asm volatile("s_wait_asynccnt 0" ::: "memory");
}

// ---------------- generic f32 -> bf16 convert ----------------
__global__ void k_f32_to_bf16(const float* __restrict__ s, unsigned short* __restrict__ d, int n) {
  int i = blockIdx.x * blockDim.x + threadIdx.x;
  if (i < n) d[i] = f2bf(s[i]);
}

// ---- weight transpose+convert: W[K,N] f32 -> Wt[Ntp,K] bf16 (rows >= N zero) ----
__global__ void k_wtrans(const float* __restrict__ W, unsigned short* __restrict__ Wt,
                         int K, int N, int Ntp) {
  int t = blockIdx.x * blockDim.x + threadIdx.x;
  if (t >= Ntp * K) return;
  int n = t / K, k = t % K;
  Wt[t] = (n < N) ? f2bf(W[(size_t)k * N + n]) : (unsigned short)0;
}

// ---------------- tiled WMMA bf16 GEMM  O = act((A@W + b)*g + be) -------------
// A: [M,K] bf16 row-major. Wt: [Ntp,K] bf16 (pre-transposed, N padded to 128).
// 256 thr = 8 waves. Tile 128x128, K-step 32, double-buffered LDS with
// global_load_async_to_lds_b128 staging when K%8==0.
#define GT_M 128
#define GT_N 128
#define GT_K 32
#define LSTR 40   // 32 + 8 pad; 80B row stride keeps all 8-elem groups 16B aligned

__global__ __launch_bounds__(256) void k_gemm_bf16(
    const unsigned short* __restrict__ A, const unsigned short* __restrict__ Wt,
    const float* __restrict__ bias, const float* __restrict__ gamma,
    const float* __restrict__ beta,
    unsigned short* __restrict__ Obf, float* __restrict__ Of32,
    int M, int N, int K, int fuse_bn_relu)
{
  __shared__ __align__(16) unsigned short As[2][GT_M * LSTR];
  __shared__ __align__(16) unsigned short Bs[2][GT_N * LSTR];

  const int tid  = threadIdx.x;
  const int wave = tid >> 5;
  const int lane = tid & 31;
  const int laneRow = lane & 15;
  const int laneHi  = lane >> 4;
  const int m0 = blockIdx.y * GT_M;
  const int n0 = blockIdx.x * GT_N;

  v8f acc[8];
#pragma unroll
  for (int c = 0; c < 8; ++c) acc[c] = v8f{0, 0, 0, 0, 0, 0, 0, 0};

  // stage one 128x32 A tile + 128x32 B tile into buffer `buf`
  auto stage = [&](int buf, int kq0) {
    bool fullK = ((K & 7) == 0) && (kq0 + GT_K <= K);
    if (fullK) {
      // 512 x 16B chunks each; per chunk: row r (0..127), segment seg (0..3)
      for (int idx = tid; idx < 512; idx += 256) {
        int r = idx >> 2, seg = idx & 3;
        async_ld_b128(A  + (size_t)(m0 + r) * K + kq0 + seg * 8,
                      &As[buf][r * LSTR + seg * 8]);
        async_ld_b128(Wt + (size_t)(n0 + r) * K + kq0 + seg * 8,
                      &Bs[buf][r * LSTR + seg * 8]);
      }
    } else {
      for (int idx = tid; idx < GT_M * GT_K; idx += 256) {
        int r = idx >> 5, kk = idx & 31;
        int gm = m0 + r, gk = kq0 + kk;
        As[buf][r * LSTR + kk] = (gm < M && gk < K) ? A[(size_t)gm * K + gk]
                                                    : (unsigned short)0;
        Bs[buf][r * LSTR + kk] = (gk < K) ? Wt[(size_t)(n0 + r) * K + gk]
                                          : (unsigned short)0;
      }
    }
  };

  int cur = 0;
  stage(0, 0);
  wait_async0();
  __syncthreads();

  for (int k0 = 0; k0 < K; k0 += GT_K) {
    int nxt = cur ^ 1;
    if (k0 + GT_K < K) stage(nxt, k0 + GT_K);   // overlap next-tile async DMA

    // A fragment per ISA 16-bit A layout:
    //  lanes 0-15 : row=lane,    K 0-7 (u[0]) and 16-23 (u[1])
    //  lanes16-31 : row=lane-16, K 8-15       and 24-31
    BfFrag af;
    const ushort8* arow = (const ushort8*)&As[cur][(wave * 16 + laneRow) * LSTR];
    af.u[0] = arow[laneHi];
    af.u[1] = arow[2 + laneHi];

#pragma unroll
    for (int c = 0; c < 8; ++c) {
      // B fragment: lanes 0-15 col K0-7,16-23; lanes 16-31 col K8-15,24-31
      BfFrag bf_;
      const ushort8* bcol = (const ushort8*)&Bs[cur][(c * 16 + laneRow) * LSTR];
      bf_.u[0] = bcol[laneHi];
      bf_.u[1] = bcol[2 + laneHi];
      acc[c] = __builtin_amdgcn_wmma_f32_16x16x32_bf16(
          false, af.bf, false, bf_.bf, (short)0, acc[c], false, false);
    }

    wait_async0();
    __syncthreads();
    cur = nxt;
  }

  // epilogue: C/D layout -> element (m0+16w + r + 8*laneHi, n0+16c+laneRow)
#pragma unroll
  for (int c = 0; c < 8; ++c) {
    int gn = n0 + c * 16 + laneRow;
    if (gn >= N) continue;
    float bb = bias  ? bias[gn]  : 0.f;
    float gg = fuse_bn_relu ? gamma[gn] : 1.f;
    float ee = fuse_bn_relu ? beta[gn]  : 0.f;
#pragma unroll
    for (int r = 0; r < 8; ++r) {
      int gm = m0 + wave * 16 + r + 8 * laneHi;
      if (gm >= M) continue;
      float v = acc[c][r] + bb;
      if (fuse_bn_relu) v = fmaxf(v * gg + ee, 0.f);
      if (Obf)  Obf[(size_t)gm * N + gn] = f2bf(v);
      if (Of32) Of32[(size_t)gm * N + gn] = v;
    }
  }
}

// ---------------- kNN: one wave32 per query point ----------------
#define KNN_MAXK 20
__global__ void k_knn(const float* __restrict__ pos, int* __restrict__ idx,
                      int n, int kk, int dil, int ksel)
{
  int wid  = blockIdx.x * (blockDim.x >> 5) + (threadIdx.x >> 5);
  int lane = threadIdx.x & 31;
  if (wid >= BB * n) return;
  int b = wid / n, i = wid % n;
  const float* base = pos + (size_t)b * n * 3;
  float pix = base[i * 3 + 0], piy = base[i * 3 + 1], piz = base[i * 3 + 2];

  float bd[KNN_MAXK]; int bi[KNN_MAXK];
  for (int t = 0; t < kk; ++t) { bd[t] = 3.4e38f; bi[t] = 0x7fffffff; }

  for (int j = lane; j < n; j += 32) {
    float dx = base[j * 3 + 0] - pix;
    float dy = base[j * 3 + 1] - piy;
    float dz = base[j * 3 + 2] - piz;
    float d2 = dx * dx + dy * dy + dz * dz;
    if (d2 < bd[kk - 1]) {
      int t = kk - 1;
      while (t > 0 && bd[t - 1] > d2) { bd[t] = bd[t - 1]; bi[t] = bi[t - 1]; --t; }
      bd[t] = d2; bi[t] = j;
    }
  }
  // cross-lane merge: kk rounds of warp argmin over lane heads
  int head = 0;
  for (int r = 0; r < kk; ++r) {
    float myd = (head < kk) ? bd[head] : 3.4e38f;
    int   myi = (head < kk) ? bi[head] : 0x7fffffff;
    float d = myd; int ii = myi;
    for (int o = 16; o; o >>= 1) {
      float od = __shfl_xor(d, o);
      int   oi = __shfl_xor(ii, o);
      if (od < d || (od == d && oi < ii)) { d = od; ii = oi; }
    }
    if (myd == d && myi == ii && head < kk) head++;
    if (lane == 0 && (r % dil) == 0)
      idx[((size_t)b * n + i) * ksel + r / dil] = ii;
  }
}

// ---------------- edge feature gather: e = [x_i, x_j - x_i] (bf16) ----------------
__global__ void k_egather(const float* __restrict__ x, const int* __restrict__ idx,
                          unsigned short* __restrict__ e, int n, int k, int C)
{
  int t = blockIdx.x * blockDim.x + threadIdx.x;
  long total = (long)BB * n * k * C;
  if (t >= total) return;
  int  c    = t % C;
  long edge = t / C;
  int  js   = (int)(edge % k);
  long ni   = edge / k;                 // b*n + i
  int  b    = (int)(ni / n);
  int  j    = idx[ni * k + js];
  float xi = x[ni * (size_t)C + c];
  float xj = x[((size_t)b * n + j) * C + c];
  e[edge * 2 * C + c]     = f2bf(xi);
  e[edge * 2 * C + C + c] = f2bf(xj - xi);
}

// ---------------- max over k neighbors (bf16 in, f32 out) ----------------
__global__ void k_edge_max(const unsigned short* __restrict__ y, float* __restrict__ xo,
                           int n, int k, int Co)
{
  int t = blockIdx.x * blockDim.x + threadIdx.x;
  if (t >= BB * n * Co) return;
  int c = t % Co;
  int node = t / Co;
  const unsigned short* yp = y + ((size_t)node * k) * Co + c;
  float m = -3.4e38f;
  for (int j = 0; j < k; ++j) m = fmaxf(m, bf2f(yp[(size_t)j * Co]));
  xo[(size_t)node * Co + c] = m;
}

// ---------------- global max pool over n rows per batch ----------------
__global__ void k_maxpool(const unsigned short* __restrict__ x, float* __restrict__ out,
                          int n, int C)
{
  int t = blockIdx.x * blockDim.x + threadIdx.x;
  if (t >= BB * C) return;
  int b = t / C, c = t % C;
  const unsigned short* xp = x + ((size_t)b * n) * C + c;
  float m = -3.4e38f;
  for (int j = 0; j < n; ++j) m = fmaxf(m, bf2f(xp[(size_t)j * C]));
  out[t] = m;
}

// ---------------- small fp32 FC with BN+ReLU (M=8 rows) ----------------
__global__ void k_fc_f32(const float* __restrict__ x, const float* __restrict__ W,
                         const float* __restrict__ bb, const float* __restrict__ g,
                         const float* __restrict__ be, float* __restrict__ out,
                         int M, int K, int N)
{
  int t = blockIdx.x * blockDim.x + threadIdx.x;
  if (t >= M * N) return;
  int m = t / N, nn = t % N;
  float acc = 0.f;
  for (int kq = 0; kq < K; ++kq) acc += x[m * K + kq] * W[kq * N + nn];
  out[t] = fmaxf((acc + bb[nn]) * g[nn] + be[nn], 0.f);
}

// ---------------- STN final 256->9 (+eye), writes ws and d_out tail ----------------
__global__ void k_stn_final(const float* __restrict__ x, const float* __restrict__ W,
                            const float* __restrict__ bb, float* __restrict__ t1,
                            float* __restrict__ out_t1)
{
  int t = blockIdx.x * blockDim.x + threadIdx.x;
  if (t >= BB * 9) return;
  int m = t / 9, j = t % 9;
  float acc = bb[j];
  for (int kq = 0; kq < 256; ++kq) acc += x[m * 256 + kq] * W[kq * 9 + j];
  if (j == 0 || j == 4 || j == 8) acc += 1.f;
  t1[t] = acc;
  out_t1[t] = acc;
}

// ---------------- pos transform: p0 = pos3 @ t1 per batch ----------------
__global__ void k_transform(const float* __restrict__ pos, const float* __restrict__ t1,
                            float* __restrict__ p0)
{
  int t = blockIdx.x * blockDim.x + threadIdx.x;
  if (t >= BB * NN * 3) return;
  int d = t % 3, p = t / 3, b = p / NN;
  const float* T = t1 + b * 9;
  p0[t] = pos[p * 3 + 0] * T[0 + d] + pos[p * 3 + 1] * T[3 + d] + pos[p * 3 + 2] * T[6 + d];
}

// ---------------- pool2: max feature pairs, mean pos pairs ----------------
__global__ void k_pool2(const float* __restrict__ x, const float* __restrict__ pos,
                        float* __restrict__ xo, float* __restrict__ poso, int n, int C)
{
  int t = blockIdx.x * blockDim.x + threadIdx.x;
  int n2 = n / 2;
  if (t < BB * n2 * C) {
    int c = t % C, node = t / C, b = node / n2, i2 = node % n2;
    size_t s = ((size_t)b * n + 2 * i2) * C + c;
    xo[t] = fmaxf(x[s], x[s + C]);
  }
  if (t < BB * n2 * 3) {
    int d = t % 3, node = t / 3, b = node / n2, i2 = node % n2;
    size_t s = ((size_t)b * n + 2 * i2) * 3 + d;
    poso[t] = 0.5f * (pos[s] + pos[s + 3]);
  }
}

// ---------------- feat = [x0, up(x1), up(x2), up(x3)] as bf16 [16384,320] -------------
__global__ void k_feat(const float* __restrict__ x0, const float* __restrict__ x1,
                       const float* __restrict__ x2, const float* __restrict__ x3,
                       unsigned short* __restrict__ feat)
{
  int t = blockIdx.x * blockDim.x + threadIdx.x;
  if (t >= BB * NN * 320) return;
  int c = t % 320, p = t / 320, b = p / NN, i = p % NN;
  float v;
  if      (c < 64)  v = x0[(size_t)p * 64 + c];
  else if (c < 128) v = x1[((size_t)b * (NN / 2) + i / 2) * 64 + (c - 64)];
  else if (c < 192) v = x2[((size_t)b * (NN / 4) + i / 4) * 64 + (c - 128)];
  else              v = x3[((size_t)b * (NN / 8) + i / 8) * 128 + (c - 192)];
  feat[t] = f2bf(v);
}

// ---------------- h = [feat, gmax, onehot] bf16 [16384,2384] ----------------
__global__ void k_hbuild(const unsigned short* __restrict__ feat,
                         const float* __restrict__ gmax, const int* __restrict__ category,
                         unsigned short* __restrict__ h)
{
  const int HC = 320 + 2048 + NCLS;
  int t = blockIdx.x * blockDim.x + threadIdx.x;
  if (t >= BB * NN * HC) return;
  int c = t % HC, p = t / HC, b = p / NN;
  unsigned short v;
  if      (c < 320)  v = feat[(size_t)p * 320 + c];
  else if (c < 2368) v = f2bf(gmax[b * 2048 + (c - 320)]);
  else               v = ((c - 2368) == category[b]) ? f2bf(1.f) : (unsigned short)0;
  h[t] = v;
}

// ---------------- final layer 128->50 + log_softmax, one wave per row -------------
__global__ void k_head_final(const unsigned short* __restrict__ x, const float* __restrict__ W,
                             const float* __restrict__ bb, float* __restrict__ out)
{
  int row  = blockIdx.x * (blockDim.x >> 5) + (threadIdx.x >> 5);
  int lane = threadIdx.x & 31;
  if (row >= BB * NN) return;
  int o0 = lane, o1 = lane + 32;
  float v0 = -3.4e38f, v1 = -3.4e38f;
  float a0 = (o0 < OUTC) ? bb[o0] : 0.f;
  float a1 = (o1 < OUTC) ? bb[o1] : 0.f;
  for (int kq = 0; kq < 128; ++kq) {
    float xv = bf2f(x[(size_t)row * 128 + kq]);
    if (o0 < OUTC) a0 += xv * W[kq * OUTC + o0];
    if (o1 < OUTC) a1 += xv * W[kq * OUTC + o1];
  }
  if (o0 < OUTC) v0 = a0;
  if (o1 < OUTC) v1 = a1;
  float m = fmaxf(v0, v1);
  for (int o = 16; o; o >>= 1) m = fmaxf(m, __shfl_xor(m, o));
  float s = 0.f;
  if (o0 < OUTC) s += __expf(v0 - m);
  if (o1 < OUTC) s += __expf(v1 - m);
  for (int o = 16; o; o >>= 1) s += __shfl_xor(s, o);
  float lse = m + __logf(s);
  if (o0 < OUTC) out[(size_t)row * OUTC + o0] = v0 - lse;
  if (o1 < OUTC) out[(size_t)row * OUTC + o1] = v1 - lse;
}

// =====================================================================
extern "C" void kernel_launch(void* const* d_in, const int* in_sizes, int n_in,
                              void* d_out, int out_size, void* d_ws, size_t ws_size,
                              hipStream_t stream)
{
  (void)in_sizes; (void)n_in; (void)out_size; (void)ws_size;

  // ---- input indices (setup_inputs dict order, nested insertion order) ----
  // 0 pos, 1 batch, 2 category,
  // stn[i]: W,b,g,be starting at 3,7,11,15,19 ; stn[5]: W=23,b=24
  // blocks[i]: 25,29,33,37 ; lin1: 41 ; head[i]: 45,49,53 ; head[3]: W=57,b=58
  auto F = [&](int i) { return (const float*)d_in[i]; };
  const float* pos = F(0);
  const int* category = (const int*)d_in[2];
  float* outF = (float*)d_out;

  // ---- bump allocator over d_ws ----
  char* wp = (char*)d_ws;
  auto alloc = [&](size_t bytes) -> void* {
    void* p = (void*)wp;
    wp += (bytes + 255) & ~(size_t)255;
    return p;
  };
  auto allocU = [&](size_t elems) { return (unsigned short*)alloc(elems * 2); };
  auto allocF = [&](size_t elems) { return (float*)alloc(elems * 4); };
  auto allocI = [&](size_t elems) { return (int*)alloc(elems * 4); };

  auto cvt = [&](const float* s, unsigned short* d, long n) {
    k_f32_to_bf16<<<dim3((unsigned)((n + 255) / 256)), 256, 0, stream>>>(s, d, (int)n);
  };
  // transpose+convert weight W[K,N] -> Wt[Ntp,K] bf16 (Ntp = N padded to 128)
  auto wtrans = [&](const float* W, int K, int N) {
    int Ntp = (N + 127) & ~127;
    unsigned short* Wt = allocU((size_t)Ntp * K);
    long tot = (long)Ntp * K;
    k_wtrans<<<dim3((unsigned)((tot + 255) / 256)), 256, 0, stream>>>(W, Wt, K, N, Ntp);
    return Wt;
  };
  auto gemm = [&](const unsigned short* A, const unsigned short* Wt,
                  const float* b, const float* g, const float* be,
                  unsigned short* obf, float* of32, int M, int N, int K) {
    dim3 grid((N + GT_N - 1) / GT_N, (M + GT_M - 1) / GT_M);
    k_gemm_bf16<<<grid, 256, 0, stream>>>(A, Wt, b, g, be, obf, of32, M, N, K, 1);
  };

  const int P = BB * NN; // 16384 points

  // ---- transpose+convert all GEMM weights (every call; no caching) ----
  unsigned short* wstn0 = wtrans(F(3),  3,    64);
  unsigned short* wstn1 = wtrans(F(7),  64,   128);
  unsigned short* wstn2 = wtrans(F(11), 128,  1024);
  unsigned short* wb0   = wtrans(F(25), 6,    64);
  unsigned short* wb1   = wtrans(F(29), 128,  64);
  unsigned short* wb2   = wtrans(F(33), 128,  64);
  unsigned short* wb3   = wtrans(F(37), 128,  128);
  unsigned short* wlin1 = wtrans(F(41), 320,  2048);
  unsigned short* wh0   = wtrans(F(45), 2384, 512);
  unsigned short* wh1   = wtrans(F(49), 512,  256);
  unsigned short* wh2   = wtrans(F(53), 256,  128);

  // ---- STN: MLP 3->64->128->1024 on raw pos (bf16 WMMA) ----
  unsigned short* posbf = allocU((size_t)P * 3);  cvt(pos, posbf, (long)P * 3);
  unsigned short* s64   = allocU((size_t)P * 64);
  unsigned short* s128  = allocU((size_t)P * 128);
  unsigned short* s1024 = allocU((size_t)P * 1024);
  gemm(posbf, wstn0, F(4),  F(5),  F(6),  s64,   nullptr, P, 64,   3);
  gemm(s64,   wstn1, F(8),  F(9),  F(10), s128,  nullptr, P, 128,  64);
  gemm(s128,  wstn2, F(12), F(13), F(14), s1024, nullptr, P, 1024, 128);

  float* smax = allocF(BB * 1024);
  k_maxpool<<<dim3((BB * 1024 + 255) / 256), 256, 0, stream>>>(s1024, smax, NN, 1024);

  float* sf1 = allocF(BB * 512);
  float* sf2 = allocF(BB * 256);
  k_fc_f32<<<dim3((BB * 512 + 255) / 256), 256, 0, stream>>>(smax, F(15), F(16), F(17), F(18), sf1, BB, 1024, 512);
  k_fc_f32<<<dim3((BB * 256 + 255) / 256), 256, 0, stream>>>(sf1,  F(19), F(20), F(21), F(22), sf2, BB, 512, 256);

  float* t1w = allocF(BB * 9);
  k_stn_final<<<1, 128, 0, stream>>>(sf2, F(23), F(24), t1w, outF + (size_t)P * OUTC);

  float* p0 = allocF((size_t)P * 3);
  k_transform<<<dim3((P * 3 + 255) / 256), 256, 0, stream>>>(pos, t1w, p0);

  // ---- EdgeConv level 0: k=20 dil=1, C 3 -> 64 ----
  int* idx0 = allocI((size_t)P * 20);
  k_knn<<<dim3((BB * NN + 7) / 8), 256, 0, stream>>>(p0, idx0, NN, 20, 1, 20);
  long E0 = (long)P * 20;
  unsigned short* e0 = allocU((size_t)E0 * 6);
  k_egather<<<dim3((unsigned)((E0 * 3 + 255) / 256)), 256, 0, stream>>>(p0, idx0, e0, NN, 20, 3);
  unsigned short* y0 = allocU((size_t)E0 * 64);
  gemm(e0, wb0, F(26), F(27), F(28), y0, nullptr, (int)E0, 64, 6);
  float* x0 = allocF((size_t)P * 64);
  k_edge_max<<<dim3((P * 64 + 255) / 256), 256, 0, stream>>>(y0, x0, NN, 20, 64);

  // ---- level 1: pool -> n=1024, k=5 dil=2, C 64 -> 64 ----
  int n1 = NN / 2;
  float* xp1 = allocF((size_t)BB * n1 * 64);
  float* p1  = allocF((size_t)BB * n1 * 3);
  k_pool2<<<dim3((BB * n1 * 64 + 255) / 256), 256, 0, stream>>>(x0, p0, xp1, p1, NN, 64);
  int* idx1 = allocI((size_t)BB * n1 * 5);
  k_knn<<<dim3((BB * n1 + 7) / 8), 256, 0, stream>>>(p1, idx1, n1, 10, 2, 5);
  long E1 = (long)BB * n1 * 5;
  unsigned short* e1 = allocU((size_t)E1 * 128);
  k_egather<<<dim3((unsigned)((E1 * 64 + 255) / 256)), 256, 0, stream>>>(xp1, idx1, e1, n1, 5, 64);
  unsigned short* y1 = allocU((size_t)E1 * 64);
  gemm(e1, wb1, F(30), F(31), F(32), y1, nullptr, (int)E1, 64, 128);
  float* x1 = allocF((size_t)BB * n1 * 64);
  k_edge_max<<<dim3((BB * n1 * 64 + 255) / 256), 256, 0, stream>>>(y1, x1, n1, 5, 64);

  // ---- level 2: n=512 ----
  int n2 = n1 / 2;
  float* xp2 = allocF((size_t)BB * n2 * 64);
  float* p2  = allocF((size_t)BB * n2 * 3);
  k_pool2<<<dim3((BB * n1 / 2 * 64 + 255) / 256), 256, 0, stream>>>(x1, p1, xp2, p2, n1, 64);
  int* idx2 = allocI((size_t)BB * n2 * 5);
  k_knn<<<dim3((BB * n2 + 7) / 8), 256, 0, stream>>>(p2, idx2, n2, 10, 2, 5);
  long E2 = (long)BB * n2 * 5;
  unsigned short* e2 = allocU((size_t)E2 * 128);
  k_egather<<<dim3((unsigned)((E2 * 64 + 255) / 256)), 256, 0, stream>>>(xp2, idx2, e2, n2, 5, 64);
  unsigned short* y2 = allocU((size_t)E2 * 64);
  gemm(e2, wb2, F(34), F(35), F(36), y2, nullptr, (int)E2, 64, 128);
  float* x2 = allocF((size_t)BB * n2 * 64);
  k_edge_max<<<dim3((BB * n2 * 64 + 255) / 256), 256, 0, stream>>>(y2, x2, n2, 5, 64);

  // ---- level 3: n=256, C 64 -> 128 ----
  int n3 = n2 / 2;
  float* xp3 = allocF((size_t)BB * n3 * 64);
  float* p3  = allocF((size_t)BB * n3 * 3);
  k_pool2<<<dim3((BB * n2 / 2 * 64 + 255) / 256), 256, 0, stream>>>(x2, p2, xp3, p3, n2, 64);
  int* idx3 = allocI((size_t)BB * n3 * 5);
  k_knn<<<dim3((BB * n3 + 7) / 8), 256, 0, stream>>>(p3, idx3, n3, 10, 2, 5);
  long E3 = (long)BB * n3 * 5;
  unsigned short* e3 = allocU((size_t)E3 * 128);
  k_egather<<<dim3((unsigned)((E3 * 64 + 255) / 256)), 256, 0, stream>>>(xp3, idx3, e3, n3, 5, 64);
  unsigned short* y3 = allocU((size_t)E3 * 128);
  gemm(e3, wb3, F(38), F(39), F(40), y3, nullptr, (int)E3, 128, 128);
  float* x3 = allocF((size_t)BB * n3 * 128);
  k_edge_max<<<dim3((BB * n3 * 128 + 255) / 256), 256, 0, stream>>>(y3, x3, n3, 5, 128);

  // ---- feat concat + lin1 + global max ----
  unsigned short* feat = allocU((size_t)P * 320);
  k_feat<<<dim3((P * 320 + 255) / 256), 256, 0, stream>>>(x0, x1, x2, x3, feat);
  unsigned short* gbuf = allocU((size_t)P * 2048);
  gemm(feat, wlin1, F(42), F(43), F(44), gbuf, nullptr, P, 2048, 320);
  float* gmax = allocF(BB * 2048);
  k_maxpool<<<dim3((BB * 2048 + 255) / 256), 256, 0, stream>>>(gbuf, gmax, NN, 2048);

  // ---- head ----
  const int HC = 320 + 2048 + NCLS;
  unsigned short* hbuf = allocU((size_t)P * HC);
  {
    long tot = (long)P * HC;
    k_hbuild<<<dim3((unsigned)((tot + 255) / 256)), 256, 0, stream>>>(feat, gmax, category, hbuf);
  }
  unsigned short* h1b = allocU((size_t)P * 512);
  unsigned short* h2b = allocU((size_t)P * 256);
  unsigned short* h3b = allocU((size_t)P * 128);
  gemm(hbuf, wh0, F(46), F(47), F(48), h1b, nullptr, P, 512, HC);
  gemm(h1b,  wh1, F(50), F(51), F(52), h2b, nullptr, P, 256, 512);
  gemm(h2b,  wh2, F(54), F(55), F(56), h3b, nullptr, P, 128, 256);

  k_head_final<<<dim3((BB * NN + 7) / 8), 256, 0, stream>>>(h3b, F(57), F(58), outF);
}
